// NeuralODE_Latent_MLP_10350871183740
// MI455X (gfx1250) — matrix operand order repaired and checked by reading, not compiled, not run on hardware
//
#include <hip/hip_runtime.h>

#define BB 2048
#define TT 128

typedef __attribute__((ext_vector_type(16))) __bf16 v16bf;
typedef __attribute__((ext_vector_type(8)))  float  v8f;

union BFragU { v16bf v; unsigned d[8]; };

__device__ __forceinline__ unsigned short f2bfh(float f){
  union { __bf16 h; unsigned short u; } c; c.h = (__bf16)f; return c.u;
}

__device__ __forceinline__ unsigned pack2bf(float a, float b){
  union { unsigned u; __bf16 h[2]; } r;
  r.h[0] = (__bf16)a; r.h[1] = (__bf16)b;
  return r.u;
}

// A-matrix 16x32 bf16 fragment from bf16 LDS (row stride in halves, even).
// ISA 7.12.2: lanes0-15 K0-7/K16-23, lanes16-31 K8-15/K24-31; K pairs share a VGPR.
__device__ __forceinline__ v16bf gatherA_bf(const unsigned short* src, int strideH, int kstep, int lane){
  int m  = lane & 15;
  int kb = ((lane >> 4) << 3) + (kstep << 5);
  const unsigned* p0 = (const unsigned*)(src + m * strideH + kb);        // k = kb .. kb+7
  const unsigned* p1 = (const unsigned*)(src + m * strideH + kb + 16);   // k = kb+16 .. kb+23
  BFragU r;
#pragma unroll
  for (int q = 0; q < 4; ++q) r.d[q] = p0[q];
#pragma unroll
  for (int q = 0; q < 4; ++q) r.d[4 + q] = p1[q];
  return r.v;
}

// B-matrix 32x16 bf16 fragment: B[k][n] = W[nglob][kglob] (x @ W^T), W bf16 [out][kstride].
// lanes0-15 K0-7/K16-23, lanes16-31 K8-15/K24-31 (same k order as A).
__device__ __forceinline__ v16bf gatherB_bf(const unsigned short* w, int strideH, int ntile, int kstep, int lane){
  int n  = (lane & 15) + (ntile << 4);
  int kb = ((lane >> 4) << 3) + (kstep << 5);
  const unsigned* p = (const unsigned*)(w + n * strideH + kb);
  BFragU r;
#pragma unroll
  for (int q = 0; q < 4; ++q) r.d[q] = p[q];
#pragma unroll
  for (int q = 0; q < 4; ++q) r.d[4 + q] = p[8 + q];
  return r.v;
}

__device__ __forceinline__ v8f splat8(float b){
  v8f c;
#pragma unroll
  for (int r = 0; r < 8; ++r) c[r] = b;
  return c;
}

__device__ __forceinline__ v8f wmma_bf(v16bf a, v16bf b, v8f c){
  return __builtin_amdgcn_wmma_f32_16x16x32_bf16(false, a, false, b, (short)0, c, false, false);
}

// C/D 16x16 f32: lane L elem r -> (m = r + 8*(L>=16), n = L&15)
__device__ __forceinline__ void storeD_bf(unsigned short* dst, int strideH, int ntile, int lane, v8f c, bool relu){
  int n  = (lane & 15) + (ntile << 4);
  int mb = (lane >> 4) << 3;
#pragma unroll
  for (int r = 0; r < 8; ++r){
    float v = c[r];
    if (relu) v = fmaxf(v, 0.0f);
    dst[(mb + r) * strideH + n] = f2bfh(v);
  }
}

// D-layout bf16 store of a full [16][32] result held as 2 n-tiles in registers
__device__ __forceinline__ void storeD2_bf(unsigned short* dst, const v8f* yt, int lane){
  int n  = lane & 15;
  int mb = (lane >> 4) << 3;
#pragma unroll
  for (int t = 0; t < 2; ++t)
#pragma unroll
    for (int r = 0; r < 8; ++r)
      dst[(mb + r) * 32 + t * 16 + n] = f2bfh(yt[t][r]);
}

// f32 weight [rows][kin] -> bf16 LDS [rows][kstride]; col kin = bias (if given), rest 0.
__device__ __forceinline__ void copyW_bf(unsigned short* dst, const float* src, const float* bias,
                                         int rows, int kin, int kstride, int tid, int nth){
  int npairs = rows * kstride / 2;
  for (int idx = tid; idx < npairs; idx += nth){
    int h0 = idx * 2;
    int r  = h0 / kstride;
    int c0 = h0 % kstride;
    float a = (c0     < kin) ? src[r * kin + c0]     : ((bias && c0     == kin) ? bias[r] : 0.0f);
    float b = (c0 + 1 < kin) ? src[r * kin + c0 + 1] : ((bias && c0 + 1 == kin) ? bias[r] : 0.0f);
    ((unsigned*)dst)[idx] = pack2bf(a, b);
  }
}

__global__ void __launch_bounds__(32) ode_kernel(
    const float* encW0, const float* encb0, const float* encW1, const float* encb1,
    const float* acW0,  const float* acb0,  const float* acW1,  const float* acb1,
    const float* dynW0, const float* dynb0, const float* dynW1, const float* dynb1,
    const float* ob, const float* acs, const float* times, float* ys)
{
  __shared__ __align__(16) unsigned short WSTB[4096];  // weight staging bf16 (<= 64x64)
  __shared__ __align__(16) unsigned short AW0B[2048];  // acW0 bf16 [64][32] (col 8 = acb0)
  __shared__ __align__(16) unsigned short AW1B[1024];  // acW1 bf16 [16][64]
  __shared__ __align__(16) unsigned short YB  [512];   // bf16 state [16][32]
  __shared__ __align__(16) unsigned short YTB [512];   // bf16 stage input [16][32]
  __shared__ __align__(16) unsigned short HIDB[1024];  // bf16 hidden [16][64]
  __shared__ __align__(16) unsigned short ACLB[512];   // bf16 action latent [16][32] (col16=1, rest 0)
  __shared__ __align__(16) unsigned short ACSB[512];   // bf16 acs tile [16][32] (col8=1, rest 0)
  __shared__ __align__(16) unsigned short XSB [1024];  // bf16 ob tile [16][64]
  __shared__ float HH[16];                             // per-row substep h

  const int lane = threadIdx.x & 31;
  const int b0   = blockIdx.x * 16;
  const int nn   = lane & 15;
  const int mb   = (lane >> 4) << 3;

  // ---- stage ob tile (bf16) + encW0 (bf16)
  for (int q = lane; q < 512; q += 32){
    int h0 = q * 2;
    int m = h0 >> 6, c = h0 & 63;
    ((unsigned*)XSB)[q] = pack2bf(ob[(size_t)(b0 + m) * 64 + c],
                                  ob[(size_t)(b0 + m) * 64 + c + 1]);
  }
  copyW_bf(WSTB, encW0, nullptr, 64, 64, 64, lane, 32);
  __syncthreads();

  // encoder layer 0
  {
    v16bf a0 = gatherA_bf(XSB, 64, 0, lane);
    v16bf a1 = gatherA_bf(XSB, 64, 1, lane);
#pragma unroll
    for (int nt = 0; nt < 4; ++nt){
      v8f c = splat8(encb0[(nt << 4) + nn]);
      c = wmma_bf(a0, gatherB_bf(WSTB, 64, nt, 0, lane), c);
      c = wmma_bf(a1, gatherB_bf(WSTB, 64, nt, 1, lane), c);
      storeD_bf(HIDB, 64, nt, lane, c, true);
    }
  }
  __syncthreads();
  copyW_bf(WSTB, encW1, nullptr, 32, 64, 64, lane, 32);
  __syncthreads();

  v8f yr[2];   // f32 state in D layout (2 n-tiles)
  {
    v16bf a0 = gatherA_bf(HIDB, 64, 0, lane);
    v16bf a1 = gatherA_bf(HIDB, 64, 1, lane);
#pragma unroll
    for (int nt = 0; nt < 2; ++nt){
      v8f c = splat8(encb1[(nt << 4) + nn]);
      c = wmma_bf(a0, gatherB_bf(WSTB, 64, nt, 0, lane), c);
      c = wmma_bf(a1, gatherB_bf(WSTB, 64, nt, 1, lane), c);
      yr[nt] = c;
    }
  }
  storeD2_bf(YB, yr, lane);
#pragma unroll
  for (int t = 0; t < 2; ++t)
#pragma unroll
    for (int r = 0; r < 8; ++r)
      ys[((size_t)(b0 + mb + r) * TT) * 32 + t * 16 + nn] = yr[t][r];

  // ---- persistent weights (bias folded into spare K column of layer-0 weights)
  copyW_bf(AW0B, acW0, acb0, 64, 8, 32, lane, 32);
  copyW_bf(AW1B, acW1, nullptr, 16, 64, 64, lane, 32);
  for (int idx = lane; idx < 256; idx += 32){
    ((unsigned*)ACSB)[idx] = 0u;
    ((unsigned*)ACLB)[idx] = 0u;
  }
  copyW_bf(WSTB, dynW0, dynb0, 64, 48, 64, lane, 32);
  __syncthreads();
  if (lane < 16){
    ACSB[lane * 32 + 8]  = 0x3F80;   // bf16 1.0 : ones column -> acb0 via weight col 8
    ACLB[lane * 32 + 16] = 0x3F80;   // bf16 1.0 : ones column (k=48) -> dynb0 via weight col 48
  }
  v16bf Bd0[2][4];
#pragma unroll
  for (int ks = 0; ks < 2; ++ks)
#pragma unroll
    for (int nt = 0; nt < 4; ++nt)
      Bd0[ks][nt] = gatherB_bf(WSTB, 64, nt, ks, lane);
  __syncthreads();
  copyW_bf(WSTB, dynW1, nullptr, 32, 64, 64, lane, 32);
  __syncthreads();
  v16bf Bd1[2][2];
#pragma unroll
  for (int ks = 0; ks < 2; ++ks)
#pragma unroll
    for (int nt = 0; nt < 2; ++nt)
      Bd1[ks][nt] = gatherB_bf(WSTB, 64, nt, ks, lane);

  float bac1 = acb1[nn];
  float bd1v[2];                      // dyn layer-1 bias, per-lane column values
  bd1v[0] = dynb1[nn];
  bd1v[1] = dynb1[16 + nn];

  v16bf aclA{};                       // A frag for K-tile 1 (acl @ k32..47, ones @ k48)

  // bias-free dynamics MLP: k = W1 relu(W0 x + b0)  (b1 added in RK combinations)
  auto feval = [&](const unsigned short* xb, v8f (&ko)[2]){
    v16bf aY = gatherA_bf(xb, 32, 0, lane);
#pragma unroll
    for (int nt = 0; nt < 4; ++nt){
      v8f c = {};                                 // inline 0 accumulator
      c = wmma_bf(aY,   Bd0[0][nt], c);
      c = wmma_bf(aclA, Bd0[1][nt], c);
      storeD_bf(HIDB, 64, nt, lane, c, true);
    }
    __syncthreads();
    v16bf h0 = gatherA_bf(HIDB, 64, 0, lane);
    v16bf h1 = gatherA_bf(HIDB, 64, 1, lane);
#pragma unroll
    for (int nt = 0; nt < 2; ++nt){
      v8f c = {};
      c = wmma_bf(h0, Bd1[0][nt], c);
      c = wmma_bf(h1, Bd1[1][nt], c);
      ko[nt] = c;
    }
    __syncthreads();
  };

  constexpr float A21 = 1.f/5.f;
  constexpr float A31 = 3.f/40.f,        A32 = 9.f/40.f;
  constexpr float A41 = 44.f/45.f,       A42 = -56.f/15.f,      A43 = 32.f/9.f;
  constexpr float A51 = 19372.f/6561.f,  A52 = -25360.f/2187.f, A53 = 64448.f/6561.f, A54 = -212.f/729.f;
  constexpr float A61 = 9017.f/3168.f,   A62 = -355.f/33.f,     A63 = 46732.f/5247.f, A64 = 49.f/176.f, A65 = -5103.f/18656.f;
  constexpr float BC1 = 35.f/384.f, BC3 = 500.f/1113.f, BC4 = 125.f/192.f, BC5 = -2187.f/6784.f, BC6 = 11.f/84.f;
  // Butcher nodes = row sums of a_ij (bias pass-through coefficients)
  constexpr float C2 = 1.f/5.f, C3 = 3.f/10.f, C4 = 4.f/5.f, C5 = 8.f/9.f, C6 = 1.f, CF = 1.f;

#pragma unroll 1
  for (int i = 0; i < TT - 1; ++i){
    if (lane < 16){
      size_t tb = (size_t)(b0 + lane) * TT + i;
      HH[lane] = (times[tb + 1] - times[tb]) * 0.5f;   // K=2 substeps
    }
#pragma unroll
    for (int q = 0; q < 2; ++q){
      int h0 = lane * 4 + q * 2;           // flat idx over [16][8]
      int m = h0 >> 3, c = h0 & 7;
      size_t g = ((size_t)(b0 + m) * TT + i) * 8 + c;
      ((unsigned*)ACSB)[(m * 32 + c) >> 1] = pack2bf(acs[g], acs[g + 1]);
    }
    __syncthreads();

    float hv[8];
#pragma unroll
    for (int r = 0; r < 8; ++r) hv[r] = HH[mb + r];

    // action latent MLP (bias of layer0 folded via ones column)
    {
      v16bf aA = gatherA_bf(ACSB, 32, 0, lane);
#pragma unroll
      for (int nt = 0; nt < 4; ++nt){
        v8f c = {};
        c = wmma_bf(aA, gatherB_bf(AW0B, 32, nt, 0, lane), c);
        storeD_bf(HIDB, 64, nt, lane, c, true);
      }
      __syncthreads();
      v16bf h0 = gatherA_bf(HIDB, 64, 0, lane);
      v16bf h1 = gatherA_bf(HIDB, 64, 1, lane);
      v8f c = splat8(bac1);
      c = wmma_bf(h0, gatherB_bf(AW1B, 64, 0, 0, lane), c);
      c = wmma_bf(h1, gatherB_bf(AW1B, 64, 0, 1, lane), c);
      storeD_bf(ACLB, 32, 0, lane, c, false);
      __syncthreads();
    }
    aclA = gatherA_bf(ACLB, 32, 0, lane);

    v8f k1[2], k2[2], k3[2], k4[2], k5[2], k6[2], yt[2];

#pragma unroll 1
    for (int j = 0; j < 2; ++j){
      feval(YB, k1);
#pragma unroll
      for (int t = 0; t < 2; ++t)
#pragma unroll
        for (int r = 0; r < 8; ++r)
          yt[t][r] = yr[t][r] + hv[r] * (A21*k1[t][r] + C2*bd1v[t]);
      storeD2_bf(YTB, yt, lane); __syncthreads();

      feval(YTB, k2);
#pragma unroll
      for (int t = 0; t < 2; ++t)
#pragma unroll
        for (int r = 0; r < 8; ++r)
          yt[t][r] = yr[t][r] + hv[r] * (A31*k1[t][r] + A32*k2[t][r] + C3*bd1v[t]);
      storeD2_bf(YTB, yt, lane); __syncthreads();

      feval(YTB, k3);
#pragma unroll
      for (int t = 0; t < 2; ++t)
#pragma unroll
        for (int r = 0; r < 8; ++r)
          yt[t][r] = yr[t][r] + hv[r] * (A41*k1[t][r] + A42*k2[t][r] + A43*k3[t][r] + C4*bd1v[t]);
      storeD2_bf(YTB, yt, lane); __syncthreads();

      feval(YTB, k4);
#pragma unroll
      for (int t = 0; t < 2; ++t)
#pragma unroll
        for (int r = 0; r < 8; ++r)
          yt[t][r] = yr[t][r] + hv[r] * (A51*k1[t][r] + A52*k2[t][r] + A53*k3[t][r] + A54*k4[t][r] + C5*bd1v[t]);
      storeD2_bf(YTB, yt, lane); __syncthreads();

      feval(YTB, k5);
#pragma unroll
      for (int t = 0; t < 2; ++t)
#pragma unroll
        for (int r = 0; r < 8; ++r)
          yt[t][r] = yr[t][r] + hv[r] * (A61*k1[t][r] + A62*k2[t][r] + A63*k3[t][r] + A64*k4[t][r] + A65*k5[t][r] + C6*bd1v[t]);
      storeD2_bf(YTB, yt, lane); __syncthreads();

      feval(YTB, k6);
#pragma unroll
      for (int t = 0; t < 2; ++t)
#pragma unroll
        for (int r = 0; r < 8; ++r)
          yr[t][r] = yr[t][r] + hv[r] * (BC1*k1[t][r] + BC3*k3[t][r] + BC4*k4[t][r] + BC5*k5[t][r] + BC6*k6[t][r] + CF*bd1v[t]);
      storeD2_bf(YB, yr, lane); __syncthreads();
    }

#pragma unroll
    for (int t = 0; t < 2; ++t)
#pragma unroll
      for (int r = 0; r < 8; ++r)
        ys[((size_t)(b0 + mb + r) * TT + (i + 1)) * 32 + t * 16 + nn] = yr[t][r];
  }
}

// ---------------- decoder: out = relu(ys @ decW0^T + b0) @ decW1^T + b1 over B*T rows ----------------
__global__ void __launch_bounds__(128) dec_kernel(
    const float* decW0, const float* decb0, const float* decW1, const float* decb1,
    const float* ys, float* out)
{
  __shared__ __align__(16) unsigned short DW0B[2048];     // bf16 [64][32]
  __shared__ __align__(16) unsigned short DW1B[4096];     // bf16 [64][64]
  __shared__ __align__(16) unsigned short XB[4][512];     // bf16 [16][32] per wave
  __shared__ __align__(16) unsigned short HB[4][1024];    // bf16 [16][64] per wave

  const int tid  = threadIdx.x;
  const int w    = tid >> 5;
  const int lane = tid & 31;
  const int nn   = lane & 15;

  copyW_bf(DW0B, decW0, nullptr, 64, 32, 32, tid, 128);
  copyW_bf(DW1B, decW1, nullptr, 64, 64, 64, tid, 128);

  const size_t rowbase = ((size_t)blockIdx.x * 4 + w) * 16;
#pragma unroll
  for (int q = 0; q < 8; ++q){
    size_t g = rowbase * 32 + lane * 16 + q * 2;
    ((unsigned*)XB[w])[lane * 8 + q] = pack2bf(ys[g], ys[g + 1]);
  }
  __syncthreads();

  v16bf aX = gatherA_bf(XB[w], 32, 0, lane);
#pragma unroll
  for (int nt = 0; nt < 4; ++nt){
    v8f c = splat8(decb0[(nt << 4) + nn]);
    c = wmma_bf(aX, gatherB_bf(DW0B, 32, nt, 0, lane), c);
    storeD_bf(HB[w], 64, nt, lane, c, true);
  }
  __syncthreads();

  v16bf h0 = gatherA_bf(HB[w], 64, 0, lane);
  v16bf h1 = gatherA_bf(HB[w], 64, 1, lane);
  const int mb = (lane >> 4) << 3;
#pragma unroll
  for (int nt = 0; nt < 4; ++nt){
    v8f c = splat8(decb1[(nt << 4) + nn]);
    c = wmma_bf(h0, gatherB_bf(DW1B, 64, nt, 0, lane), c);
    c = wmma_bf(h1, gatherB_bf(DW1B, 64, nt, 1, lane), c);
#pragma unroll
    for (int r = 0; r < 8; ++r)
      out[(rowbase + mb + r) * 64 + (nt << 4) + nn] = c[r];
  }
}

extern "C" void kernel_launch(void* const* d_in, const int* in_sizes, int n_in,
                              void* d_out, int out_size, void* d_ws, size_t ws_size,
                              hipStream_t stream) {
  (void)in_sizes; (void)n_in; (void)out_size; (void)ws_size;
  const float* encW0 = (const float*)d_in[0];
  const float* encb0 = (const float*)d_in[1];
  const float* encW1 = (const float*)d_in[2];
  const float* encb1 = (const float*)d_in[3];
  const float* acW0  = (const float*)d_in[4];
  const float* acb0  = (const float*)d_in[5];
  const float* acW1  = (const float*)d_in[6];
  const float* acb1  = (const float*)d_in[7];
  const float* dynW0 = (const float*)d_in[8];
  const float* dynb0 = (const float*)d_in[9];
  const float* dynW1 = (const float*)d_in[10];
  const float* dynb1 = (const float*)d_in[11];
  const float* decW0 = (const float*)d_in[12];
  const float* decb0 = (const float*)d_in[13];
  const float* decW1 = (const float*)d_in[14];
  const float* decb1 = (const float*)d_in[15];
  const float* ob    = (const float*)d_in[16];
  const float* acs   = (const float*)d_in[17];
  const float* times = (const float*)d_in[18];

  float* ys = (float*)d_ws;   // B*T*32 f32 = 32 MB scratch latent trajectory

  ode_kernel<<<BB / 16, 32, 0, stream>>>(
      encW0, encb0, encW1, encb1, acW0, acb0, acW1, acb1,
      dynW0, dynb0, dynW1, dynb1, ob, acs, times, ys);

  dec_kernel<<<(BB * TT) / 64, 128, 0, stream>>>(
      decW0, decb0, decW1, decb1, ys, (float*)d_out);
}
